// VAE_61469571940629
// MI455X (gfx1250) — matrix operand order repaired and checked by reading, not compiled, not run on hardware
//
#include <hip/hip_runtime.h>
#include <hip/hip_bf16.h>

// ---------------------------------------------------------------------------
// BitNet-1.58 VAE forward on MI455X (gfx1250).
// Big layers: V_WMMA_F32_16X16X32_F16, ternary weights exact in f16,
// per-layer scale folded into the f32 epilogue.
// ---------------------------------------------------------------------------

typedef __attribute__((ext_vector_type(16))) _Float16 v16h;
typedef __attribute__((ext_vector_type(8)))  float    v8f;

#define EPSQ 1e-5f

// ---------------- small helpers ----------------

__global__ void zero_kernel(float* p, int n) {
    int i = threadIdx.x;
    if (i < n) p[i] = 0.f;
}

// sum(|W|) -> accum (atomic)
__global__ __launch_bounds__(256)
void abssum_kernel(const float* __restrict__ W, int n, float* __restrict__ accum) {
    __shared__ float red[256];
    int tid = threadIdx.x;
    float s = 0.f;
    for (int i = blockIdx.x * 256 + tid; i < n; i += gridDim.x * 256)
        s += fabsf(W[i]);
    red[tid] = s;
    __syncthreads();
    for (int st = 128; st > 0; st >>= 1) {
        if (tid < st) red[tid] += red[tid + st];
        __syncthreads();
    }
    if (tid == 0) atomicAdd(accum, red[0]);
}

__device__ __forceinline__ float ternary(float w, float beta) {
    return fminf(fmaxf(rintf(w * beta), -1.f), 1.f);
}

// W (N x K row-major, f32) -> Wq (Np x Kp, f16 ternary {-1,0,1}, zero padded)
__global__ __launch_bounds__(256)
void quantize_kernel(const float* __restrict__ W, _Float16* __restrict__ Wq,
                     int N, int K, int Np, int Kp,
                     const float* __restrict__ sumAbs, float invCount) {
    int idx = blockIdx.x * 256 + threadIdx.x;
    if (idx >= Np * Kp) return;
    int n = idx / Kp, k = idx % Kp;
    float t = 0.f;
    if (n < N && k < K) {
        float s = fmaxf(sumAbs[0] * invCount, EPSQ);
        t = ternary(W[(size_t)n * K + k], 1.f / s);
    }
    Wq[(size_t)n * Kp + k] = (_Float16)t;
}

// x (B x K f32) -> X (B x Kp f16, zero padded)
__global__ __launch_bounds__(256)
void pack_x_kernel(const float* __restrict__ x, _Float16* __restrict__ X,
                   int B, int K, int Kp) {
    long long idx = (long long)blockIdx.x * 256 + threadIdx.x;
    if (idx >= (long long)B * Kp) return;
    int row = (int)(idx / Kp), k = (int)(idx % Kp);
    float v = (k < K) ? x[(size_t)row * K + k] : 0.f;
    X[(size_t)row * Kp + k] = (_Float16)v;
}

// ---------------- WMMA GEMM ----------------
// Out[m,n] = act( s * sum_k A[m,k] * Wq[n,k] )
// Block: 256 threads = 8 waves, tile 128(M) x 64(N), wave = 32x32 = 2x2 wmma frags.
// A: B x Kp f16 row-major.  Wq: Np x Kp f16 (ternary, zero-padded rows/cols).
// actMode 0: leaky-relu, store f16 (stride outStride, guard n < Nstore)
// actMode 1: sigmoid,    store f32 to OutF

union Frag { uint4 u[2]; v16h h; };

__global__ __launch_bounds__(256)
void bitgemm_wmma(const _Float16* __restrict__ A, const _Float16* __restrict__ Wq,
                  _Float16* __restrict__ OutH, float* __restrict__ OutF,
                  const float* __restrict__ sumAbs, float invCount,
                  int Kp, int outStride, int Nstore, int actMode) {
    const int lane  = threadIdx.x & 31;
    const int wave  = threadIdx.x >> 5;
    const int waveM = wave >> 1;          // 0..3
    const int waveN = wave & 1;           // 0..1
    const int l15   = lane & 15;
    const int hgr   = lane >> 4;          // lane half-group

    const size_t mWave = (size_t)blockIdx.x * 128 + waveM * 32;
    const int    nWave = blockIdx.y * 64 + waveN * 32;

    // A fragment (16x32 f16): lane l15 = M row; elements 0..7 -> K = hgr*8..,
    // elements 8..15 -> K = 16 + hgr*8..  (two b128 loads per frag)
    const _Float16* aRow0 = A + (mWave + l15) * (size_t)Kp + hgr * 8;
    const _Float16* aRow1 = aRow0 + (size_t)16 * Kp;
    // B fragment (32x16 f16): lane l15 = N col; elements 0..7 -> K = hgr*8..,
    // elements 8..15 -> K = 16 + hgr*8..  (two b128 loads per frag)
    const _Float16* bCol0 = Wq + (size_t)(nWave + l15) * Kp + hgr * 8;
    const _Float16* bCol1 = bCol0 + (size_t)16 * Kp;

    v8f c[2][2] = {};

    for (int k0 = 0; k0 < Kp; k0 += 32) {
        Frag a0, a1, b0, b1;
        a0.u[0] = *(const uint4*)(aRow0 + k0);
        a0.u[1] = *(const uint4*)(aRow0 + k0 + 16);
        a1.u[0] = *(const uint4*)(aRow1 + k0);
        a1.u[1] = *(const uint4*)(aRow1 + k0 + 16);
        b0.u[0] = *(const uint4*)(bCol0 + k0);
        b0.u[1] = *(const uint4*)(bCol0 + k0 + 16);
        b1.u[0] = *(const uint4*)(bCol1 + k0);
        b1.u[1] = *(const uint4*)(bCol1 + k0 + 16);

        c[0][0] = __builtin_amdgcn_wmma_f32_16x16x32_f16(false, a0.h, false, b0.h, (short)0, c[0][0], false, false);
        c[0][1] = __builtin_amdgcn_wmma_f32_16x16x32_f16(false, a0.h, false, b1.h, (short)0, c[0][1], false, false);
        c[1][0] = __builtin_amdgcn_wmma_f32_16x16x32_f16(false, a1.h, false, b0.h, (short)0, c[1][0], false, false);
        c[1][1] = __builtin_amdgcn_wmma_f32_16x16x32_f16(false, a1.h, false, b1.h, (short)0, c[1][1], false, false);
    }

    const float s = fmaxf(sumAbs[0] * invCount, EPSQ);

    // C layout: lane l15 = N; VGPR r -> M = r (+8 for lane group 1)
    for (int i = 0; i < 2; ++i) {
        size_t mBase = mWave + i * 16 + hgr * 8;
        for (int j = 0; j < 2; ++j) {
            int nOut = nWave + j * 16 + l15;
            if (nOut >= Nstore) continue;
            for (int r = 0; r < 8; ++r) {
                float v = c[i][j][r] * s;
                size_t m = mBase + r;
                if (actMode == 0) {
                    v = (v >= 0.f) ? v : 0.2f * v;
                    OutH[m * (size_t)outStride + nOut] = (_Float16)v;
                } else {
                    v = 1.f / (1.f + expf(-v));
                    OutF[m * (size_t)outStride + nOut] = v;
                }
            }
        }
    }
}

// ---------------- tiny layers (N=2 or K=2): VALU ----------------

// mean/logvar heads (N=2 each, K=200) + reparameterize z = mu + lv*eps
__global__ __launch_bounds__(256)
void head_kernel(const _Float16* __restrict__ h3, int Kp,
                 const float* __restrict__ Wmu, const float* __restrict__ Wlv,
                 const float* __restrict__ sumMu, const float* __restrict__ sumLv,
                 float invCount, const float* __restrict__ eps,
                 float* __restrict__ outMean, float* __restrict__ outLv,
                 float* __restrict__ z, int B, int L) {
    __shared__ float qmu[2 * 200];
    __shared__ float qlv[2 * 200];
    float sMu = fmaxf(sumMu[0] * invCount, EPSQ);
    float sLv = fmaxf(sumLv[0] * invCount, EPSQ);
    for (int i = threadIdx.x; i < 2 * L; i += 256) {
        qmu[i] = ternary(Wmu[i], 1.f / sMu);
        qlv[i] = ternary(Wlv[i], 1.f / sLv);
    }
    __syncthreads();
    int row = blockIdx.x * 256 + threadIdx.x;
    if (row >= B) return;
    const _Float16* hp = h3 + (size_t)row * Kp;
    float a0 = 0.f, a1 = 0.f, a2 = 0.f, a3 = 0.f;
    for (int k = 0; k < L; ++k) {
        float hv = (float)hp[k];
        a0 += hv * qmu[k];
        a1 += hv * qmu[L + k];
        a2 += hv * qlv[k];
        a3 += hv * qlv[L + k];
    }
    float mu0 = a0 * sMu, mu1 = a1 * sMu, lv0 = a2 * sLv, lv1 = a3 * sLv;
    outMean[(size_t)row * 2]     = mu0;
    outMean[(size_t)row * 2 + 1] = mu1;
    outLv[(size_t)row * 2]       = lv0;
    outLv[(size_t)row * 2 + 1]   = lv1;
    float e0 = eps[(size_t)row * 2], e1 = eps[(size_t)row * 2 + 1];
    z[(size_t)row * 2]     = mu0 + lv0 * e0;
    z[(size_t)row * 2 + 1] = mu1 + lv1 * e1;
}

// d1: out(B x 200, padded 224) = lrelu(s * (z @ tern(W_d1)^T)), K = 2
__global__ __launch_bounds__(256)
void d1_kernel(const float* __restrict__ z, const float* __restrict__ Wd1,
               const float* __restrict__ sumD1, float invCount,
               _Float16* __restrict__ out, int B, int L, int Kp) {
    int idx = blockIdx.x * 256 + threadIdx.x;
    if (idx >= B * Kp) return;
    int row = idx / Kp, n = idx % Kp;
    float v = 0.f;
    if (n < L) {
        float s = fmaxf(sumD1[0] * invCount, EPSQ);
        float b = 1.f / s;
        float w0 = ternary(Wd1[n * 2], b);
        float w1 = ternary(Wd1[n * 2 + 1], b);
        v = (z[(size_t)row * 2] * w0 + z[(size_t)row * 2 + 1] * w1) * s;
        v = (v >= 0.f) ? v : 0.2f * v;
    }
    out[(size_t)row * Kp + n] = (_Float16)v;
}

// ---------------- orchestration ----------------

extern "C" void kernel_launch(void* const* d_in, const int* in_sizes, int n_in,
                              void* d_out, int out_size, void* d_ws, size_t ws_size,
                              hipStream_t stream) {
    (void)in_sizes; (void)n_in; (void)out_size; (void)ws_size;
    const int B = 65536, D = 784, H = 400, L = 200;
    const int KpD = 800, KpH = 416, KpL = 224;   // K padded to 32
    const int NpH = 448, NpL = 256, NpD = 832;   // N padded to 64

    const float* x   = (const float*)d_in[0];
    const float* eps = (const float*)d_in[1];
    const float* We1 = (const float*)d_in[2];
    const float* We2 = (const float*)d_in[3];
    const float* We3 = (const float*)d_in[4];
    const float* Wmu = (const float*)d_in[5];
    const float* Wlv = (const float*)d_in[6];
    const float* Wd1 = (const float*)d_in[7];
    const float* Wd2 = (const float*)d_in[8];
    const float* Wd3 = (const float*)d_in[9];
    const float* Wd4 = (const float*)d_in[10];
    const float* Wd5 = (const float*)d_in[11];

    float* out_xhat = (float*)d_out;
    float* out_mean = out_xhat + (size_t)B * D;
    float* out_lv   = out_mean + (size_t)B * 2;

    // ws suballocation
    char* base = (char*)d_ws;
    size_t off = 0;
    auto alloc = [&](size_t bytes) -> char* {
        char* p = base + off;
        off = (off + bytes + 255) & ~(size_t)255;
        return p;
    };
    float*    sums = (float*)alloc(16 * sizeof(float));
    _Float16* qe1  = (_Float16*)alloc((size_t)NpH * KpD * 2);
    _Float16* qe2  = (_Float16*)alloc((size_t)NpH * KpH * 2);
    _Float16* qe3  = (_Float16*)alloc((size_t)NpL * KpH * 2);
    _Float16* qd2  = (_Float16*)alloc((size_t)NpH * KpL * 2);
    _Float16* qd3  = (_Float16*)alloc((size_t)NpH * KpH * 2);
    _Float16* qd4  = (_Float16*)alloc((size_t)NpH * KpH * 2);
    _Float16* qd5  = (_Float16*)alloc((size_t)NpD * KpH * 2);
    _Float16* X    = (_Float16*)alloc((size_t)B * KpD * 2);  // 105 MB
    _Float16* P    = (_Float16*)alloc((size_t)B * KpH * 2);  // 54.5 MB
    float*    zbuf = (float*)alloc((size_t)B * 2 * 4);
    _Float16* Q    = X;  // Q (<=54.5 MB) aliases X once x-f16 is consumed

    zero_kernel<<<1, 32, 0, stream>>>(sums, 16);

    auto red = [&](const float* Wp, int n, int slot) {
        int blocks = (n + 255) / 256;
        if (blocks > 512) blocks = 512;
        abssum_kernel<<<blocks, 256, 0, stream>>>(Wp, n, sums + slot);
    };
    red(We1, H * D, 0); red(We2, H * H, 1); red(We3, L * H, 2);
    red(Wmu, 2 * L, 3); red(Wlv, 2 * L, 4); red(Wd1, L * 2, 5);
    red(Wd2, H * L, 6); red(Wd3, H * H, 7); red(Wd4, H * H, 8);
    red(Wd5, D * H, 9);

    auto quant = [&](const float* Wp, _Float16* Wq, int N, int K, int Np, int Kp,
                     int slot, float inv) {
        int total = Np * Kp;
        quantize_kernel<<<(total + 255) / 256, 256, 0, stream>>>(Wp, Wq, N, K, Np, Kp,
                                                                 sums + slot, inv);
    };
    quant(We1, qe1, H, D, NpH, KpD, 0, 1.f / (H * D));
    quant(We2, qe2, H, H, NpH, KpH, 1, 1.f / (H * H));
    quant(We3, qe3, L, H, NpL, KpH, 2, 1.f / (L * H));
    quant(Wd2, qd2, H, L, NpH, KpL, 6, 1.f / (H * L));
    quant(Wd3, qd3, H, H, NpH, KpH, 7, 1.f / (H * H));
    quant(Wd4, qd4, H, H, NpH, KpH, 8, 1.f / (H * H));
    quant(Wd5, qd5, D, H, NpD, KpH, 9, 1.f / (D * H));

    {
        long long total = (long long)B * KpD;
        pack_x_kernel<<<(unsigned)((total + 255) / 256), 256, 0, stream>>>(x, X, B, D, KpD);
    }

    auto gemm = [&](const _Float16* Ain, const _Float16* Wq, _Float16* OutH,
                    float* OutF, int Kp, int Np, int outStride, int Nstore,
                    int slot, float inv, int act) {
        dim3 grid(B / 128, Np / 64);
        bitgemm_wmma<<<grid, 256, 0, stream>>>(Ain, Wq, OutH, OutF, sums + slot, inv,
                                               Kp, outStride, Nstore, act);
    };

    // encoder
    gemm(X, qe1, P, nullptr, KpD, NpH, KpH, KpH, 0, 1.f / (H * D), 0);  // x -> h1 (P)
    gemm(P, qe2, Q, nullptr, KpH, NpH, KpH, KpH, 1, 1.f / (H * H), 0);  // h1 -> h2 (Q)
    gemm(Q, qe3, P, nullptr, KpH, NpL, KpL, KpL, 2, 1.f / (L * H), 0);  // h2 -> h3 (P)
    // heads + reparameterize
    head_kernel<<<B / 256, 256, 0, stream>>>(P, KpL, Wmu, Wlv, sums + 3, sums + 4,
                                             1.f / (2 * L), eps, out_mean, out_lv,
                                             zbuf, B, L);
    // decoder
    d1_kernel<<<(B * KpL + 255) / 256, 256, 0, stream>>>(zbuf, Wd1, sums + 5,
                                                         1.f / (L * 2), Q, B, L, KpL);
    gemm(Q, qd2, P, nullptr, KpL, NpH, KpH, KpH, 6, 1.f / (H * L), 0);
    gemm(P, qd3, Q, nullptr, KpH, NpH, KpH, KpH, 7, 1.f / (H * H), 0);
    gemm(Q, qd4, P, nullptr, KpH, NpH, KpH, KpH, 8, 1.f / (H * H), 0);
    gemm(P, qd5, nullptr, out_xhat, KpH, NpD, D, D, 9, 1.f / (D * H), 1);  // sigmoid f32
}
